// GraphDecoder_20323785244760
// MI455X (gfx1250) — matrix operand order repaired and checked by reading, not compiled, not run on hardware
//
#include <hip/hip_runtime.h>
#include <hip/hip_bf16.h>

#define Hdim   768
#define Odim   72
#define Bdim   256
#define Tsteps 100
#define SRCLEN 16
#define Gdim   (4 * Hdim)          // 3072 gate columns
#define NWG    64
#define COLS_PER_WG (Gdim / NWG)   // 48
#define THREADS 256
#define PADK   (Hdim + 8)          // padded LDS row: 776 el = 1552 B (bank-conflict-free)

// dynamic LDS: weight slice (48 x PADK bf16) + fp32 h tile (4 x Hdim)
#define WSLICE_BYTES (COLS_PER_WG * PADK * 2)              // 74,496
#define SMEM_BYTES   (WSLICE_BYTES + 4 * Hdim * 4)         // 86,784

typedef __attribute__((ext_vector_type(16))) __bf16 v16bf;
typedef __attribute__((ext_vector_type(8)))  float  v8f;

union U16bf { v16bf v; uint4 q[2]; };

// ---------------------------------------------------------------------------
// Prologue A: bf16 weight images + combined recurrent weight Wc = W_ih + W_hh
// ---------------------------------------------------------------------------
__global__ void prep_weights(const float* __restrict__ wih,
                             const float* __restrict__ whh,
                             __bf16* __restrict__ wc_b,
                             __bf16* __restrict__ wih_b,
                             __bf16* __restrict__ whh_b) {
    int i = blockIdx.x * blockDim.x + threadIdx.x;
    int stride = gridDim.x * blockDim.x;
    for (; i < Gdim * Hdim; i += stride) {
        float a = wih[i], b = whh[i];
        wih_b[i] = (__bf16)a;
        whh_b[i] = (__bf16)b;
        wc_b[i]  = (__bf16)(a + b);
    }
}

// ---------------------------------------------------------------------------
// Prologue B: bf16 x0 = src[:, -1], bf16 h0, zero the grid-barrier counter
// (counter reset every launch -> deterministic across graph replays)
// ---------------------------------------------------------------------------
__global__ void prep_state(const float* __restrict__ src,
                           const float* __restrict__ h0,
                           __bf16* __restrict__ xbuf,
                           __bf16* __restrict__ hbuf,
                           unsigned* __restrict__ counter) {
    int i = blockIdx.x * blockDim.x + threadIdx.x;
    if (i == 0) *counter = 0u;
    int stride = gridDim.x * blockDim.x;
    for (; i < Bdim * Hdim; i += stride) {
        int b = i / Hdim, k = i % Hdim;
        xbuf[i] = (__bf16)src[((size_t)b * SRCLEN + (SRCLEN - 1)) * Hdim + k];
        hbuf[i] = (__bf16)h0[i];
    }
}

// ---------------------------------------------------------------------------
// Device helpers
// ---------------------------------------------------------------------------
__device__ __forceinline__ float sigmoid_f(float x) {
    return 1.0f / (1.0f + __expf(-x));
}
__device__ __forceinline__ float tanh_f(float x) {
    return 1.0f - 2.0f / (__expf(2.0f * x) + 1.0f);
}

// Grid-wide split barrier, device (agent) scope.
__device__ __forceinline__ void grid_sync(unsigned* cnt, unsigned target) {
    __builtin_amdgcn_fence(__ATOMIC_RELEASE, "agent");   // flush dirty data
    __syncthreads();
    if (threadIdx.x == 0) {
        __hip_atomic_fetch_add(cnt, 1u, __ATOMIC_ACQ_REL, __HIP_MEMORY_SCOPE_AGENT);
        while (__hip_atomic_load(cnt, __ATOMIC_ACQUIRE, __HIP_MEMORY_SCOPE_AGENT) < target) {
            __builtin_amdgcn_s_sleep(2);
        }
    }
    __syncthreads();
    __builtin_amdgcn_fence(__ATOMIC_ACQUIRE, "agent");   // invalidate stale lines
}

// ---------------------------------------------------------------------------
// K-loop, B operand direct from global (used only for step 0's two GEMMs).
// A: (256 x 768) row-major bf16.  W: (3072 x 768) row-major bf16 (B = W^T).
// ---------------------------------------------------------------------------
__device__ __forceinline__ void wmma_kloop(const __bf16* __restrict__ A,
                                           const __bf16* __restrict__ W,
                                           int m0, int colBase, int lane,
                                           v8f (&acc)[2][3]) {
    const int mlo = lane & 15;
    const int grp = (lane >> 4) & 1;
    const __bf16* a0 = A + (size_t)(m0 + mlo) * Hdim;
    const __bf16* a1 = a0 + (size_t)16 * Hdim;
    const __bf16* b0 = W + (size_t)(colBase + 0 * 16 + mlo) * Hdim + grp * 16;
    const __bf16* b1 = W + (size_t)(colBase + 1 * 16 + mlo) * Hdim + grp * 16;
    const __bf16* b2 = W + (size_t)(colBase + 2 * 16 + mlo) * Hdim + grp * 16;

    for (int k = 0; k < Hdim; k += 32) {
        U16bf Am0, Am1, Bw0, Bw1, Bw2;
        Am0.q[0] = *(const uint4*)(a0 + k + grp * 16);
        Am0.q[1] = *(const uint4*)(a0 + k + 8 + grp * 16);
        Am1.q[0] = *(const uint4*)(a1 + k + grp * 16);
        Am1.q[1] = *(const uint4*)(a1 + k + 8 + grp * 16);
        const uint4* p0 = (const uint4*)(b0 + k); Bw0.q[0] = p0[0]; Bw0.q[1] = p0[1];
        const uint4* p1 = (const uint4*)(b1 + k); Bw1.q[0] = p1[0]; Bw1.q[1] = p1[1];
        const uint4* p2 = (const uint4*)(b2 + k); Bw2.q[0] = p2[0]; Bw2.q[1] = p2[1];
        __builtin_prefetch(b0 + k + 128, 0, 1);

        acc[0][0] = __builtin_amdgcn_wmma_f32_16x16x32_bf16(false, Am0.v, false, Bw0.v, (short)0, acc[0][0], false, false);
        acc[0][1] = __builtin_amdgcn_wmma_f32_16x16x32_bf16(false, Am0.v, false, Bw1.v, (short)0, acc[0][1], false, false);
        acc[0][2] = __builtin_amdgcn_wmma_f32_16x16x32_bf16(false, Am0.v, false, Bw2.v, (short)0, acc[0][2], false, false);
        acc[1][0] = __builtin_amdgcn_wmma_f32_16x16x32_bf16(false, Am1.v, false, Bw0.v, (short)0, acc[1][0], false, false);
        acc[1][1] = __builtin_amdgcn_wmma_f32_16x16x32_bf16(false, Am1.v, false, Bw1.v, (short)0, acc[1][1], false, false);
        acc[1][2] = __builtin_amdgcn_wmma_f32_16x16x32_bf16(false, Am1.v, false, Bw2.v, (short)0, acc[1][2], false, false);
    }
}

// ---------------------------------------------------------------------------
// K-loop, B operand from LDS-staged weight slice (steps 1..99).
// wsl rows are padded to PADK elements: row stride 388 DWORDs = 4 mod 64 banks,
// so the per-lane 16B ds_load pairs hit every LDS bank exactly twice per wave.
// ---------------------------------------------------------------------------
__device__ __forceinline__ void wmma_kloop_lds(const __bf16* __restrict__ A,
                                               const __bf16* __restrict__ wsl,
                                               int m0, int lane,
                                               v8f (&acc)[2][3]) {
    const int mlo = lane & 15;
    const int grp = (lane >> 4) & 1;
    const __bf16* a0 = A + (size_t)(m0 + mlo) * Hdim;
    const __bf16* a1 = a0 + (size_t)16 * Hdim;
    const __bf16* b0 = wsl + (0 * 16 + mlo) * PADK + grp * 16;
    const __bf16* b1 = wsl + (1 * 16 + mlo) * PADK + grp * 16;
    const __bf16* b2 = wsl + (2 * 16 + mlo) * PADK + grp * 16;

    for (int k = 0; k < Hdim; k += 32) {
        U16bf Am0, Am1, Bw0, Bw1, Bw2;
        Am0.q[0] = *(const uint4*)(a0 + k + grp * 16);
        Am0.q[1] = *(const uint4*)(a0 + k + 8 + grp * 16);
        Am1.q[0] = *(const uint4*)(a1 + k + grp * 16);
        Am1.q[1] = *(const uint4*)(a1 + k + 8 + grp * 16);
        Bw0.q[0] = *(const uint4*)(b0 + k); Bw0.q[1] = *(const uint4*)(b0 + k + 8);
        Bw1.q[0] = *(const uint4*)(b1 + k); Bw1.q[1] = *(const uint4*)(b1 + k + 8);
        Bw2.q[0] = *(const uint4*)(b2 + k); Bw2.q[1] = *(const uint4*)(b2 + k + 8);

        acc[0][0] = __builtin_amdgcn_wmma_f32_16x16x32_bf16(false, Am0.v, false, Bw0.v, (short)0, acc[0][0], false, false);
        acc[0][1] = __builtin_amdgcn_wmma_f32_16x16x32_bf16(false, Am0.v, false, Bw1.v, (short)0, acc[0][1], false, false);
        acc[0][2] = __builtin_amdgcn_wmma_f32_16x16x32_bf16(false, Am0.v, false, Bw2.v, (short)0, acc[0][2], false, false);
        acc[1][0] = __builtin_amdgcn_wmma_f32_16x16x32_bf16(false, Am1.v, false, Bw0.v, (short)0, acc[1][0], false, false);
        acc[1][1] = __builtin_amdgcn_wmma_f32_16x16x32_bf16(false, Am1.v, false, Bw1.v, (short)0, acc[1][1], false, false);
        acc[1][2] = __builtin_amdgcn_wmma_f32_16x16x32_bf16(false, Am1.v, false, Bw2.v, (short)0, acc[1][2], false, false);
    }
}

// ---------------------------------------------------------------------------
// Persistent LSTM decoder: 64 WGs x 256 threads, 100 steps, 2 grid syncs/step.
// Wc weight slice staged in LDS once for all 99 recurrent steps.
// ---------------------------------------------------------------------------
__global__ void __launch_bounds__(THREADS, 1)
lstm_decoder(const __bf16* __restrict__ wc_b,
             const __bf16* __restrict__ wih_b,
             const __bf16* __restrict__ whh_b,
             const __bf16* __restrict__ xbuf,
             __bf16* __restrict__ hbuf,
             float* __restrict__ gates,
             unsigned* __restrict__ counter,
             const float* __restrict__ c0,
             const float* __restrict__ b_ih,
             const float* __restrict__ b_hh,
             const float* __restrict__ w_post,
             const float* __restrict__ b_post,
             float* __restrict__ out) {
    extern __shared__ char smem[];
    __bf16* wslice = (__bf16*)smem;                         // 48 x PADK bf16
    float*  lds_h  = (float*)(smem + WSLICE_BYTES);         // 4 x Hdim fp32

    const int wg   = blockIdx.x;
    const int tid  = threadIdx.x;
    const int wave = tid >> 5;
    const int lane = tid & 31;
    const int mlo  = lane & 15;
    const int grp  = (lane >> 4) & 1;
    const int colBase = wg * COLS_PER_WG;  // 48 gate columns per WG
    const int m0      = wave * 32;         // 2 M-tiles per wave

    // phase-2 ownership: WG owns 4 batch rows; c lives in registers all 100 steps
    const int bl = tid >> 6;               // 0..3
    const int b  = wg * 4 + bl;
    const int j0 = (tid & 63) * 12;        // 12 hidden cols per thread
    float creg[12];
    for (int u = 0; u < 12; ++u) creg[u] = c0[(size_t)b * Hdim + j0 + u];

    // ---- stage this WG's Wc slice (48 x 768 bf16) into padded LDS, once ----
    {
        const uint4* wsrc = (const uint4*)(wc_b + (size_t)colBase * Hdim);
        for (int q = tid; q < COLS_PER_WG * (Hdim / 8); q += THREADS) {
            int r = q / (Hdim / 8);
            int c = q % (Hdim / 8);
            ((uint4*)(wslice + r * PADK))[c] = wsrc[r * (Hdim / 8) + c];
        }
    }
    __syncthreads();

    unsigned bar = 0;
    for (int t = 0; t < Tsteps; ++t) {
        // ---------------- phase 1: gates = x@Wih^T + h@Whh^T + bias ----------
        v8f acc[2][3];
        for (int j = 0; j < 3; ++j) {
            int n = colBase + j * 16 + mlo;
            float bn = b_ih[n] + b_hh[n];
            v8f a;
            for (int r = 0; r < 8; ++r) a[r] = bn;
            acc[0][j] = a; acc[1][j] = a;
        }
        if (t == 0) {
            wmma_kloop(xbuf, wih_b, m0, colBase, lane, acc);   // x0 @ W_ih^T
            wmma_kloop(hbuf, whh_b, m0, colBase, lane, acc);   // h0 @ W_hh^T
        } else {
            wmma_kloop_lds(hbuf, wslice, m0, lane, acc);       // h @ Wc^T, B from LDS
        }
        // C/D layout: VGPR r -> M = r + 8*(lane/16), N = lane%16
        for (int i = 0; i < 2; ++i)
            for (int j = 0; j < 3; ++j)
                for (int r = 0; r < 8; ++r) {
                    int row = m0 + i * 16 + grp * 8 + r;
                    int col = colBase + j * 16 + mlo;
                    gates[(size_t)row * Gdim + col] = acc[i][j][r];
                }
        ++bar; grid_sync(counter, bar * NWG);

        // ---------------- phase 2: elementwise LSTM + fused output proj ------
        const size_t gb = (size_t)b * Gdim;
        float hv[12];
        for (int u = 0; u < 12; ++u) {
            int j = j0 + u;
            float ig = gates[gb + j];
            float fg = gates[gb + Hdim + j];
            float gg = gates[gb + 2 * Hdim + j];
            float og = gates[gb + 3 * Hdim + j];
            float cc = sigmoid_f(fg) * creg[u] + sigmoid_f(ig) * tanh_f(gg);
            creg[u] = cc;
            hv[u] = sigmoid_f(og) * tanh_f(cc);
        }
        for (int u = 0; u < 12; ++u) {
            int j = j0 + u;
            hbuf[(size_t)b * Hdim + j] = (__bf16)hv[u];  // next step's WMMA A operand
            lds_h[bl * Hdim + j] = hv[u];                // fp32 for local projection
        }
        __syncthreads();
        // out[b, t, :] = h @ W_post^T + b_post   (4 rows x 72 outs per WG)
        for (int idx = tid; idx < 4 * Odim; idx += THREADS) {
            int pb = idx / Odim, o = idx % Odim;
            float s = b_post[o];
            const float* hr = &lds_h[pb * Hdim];
            const float* wr = w_post + (size_t)o * Hdim;
            #pragma unroll 4
            for (int j = 0; j < Hdim; ++j) s = fmaf(hr[j], wr[j], s);
            out[((size_t)(wg * 4 + pb) * Tsteps + t) * Odim + o] = s;
        }
        ++bar; grid_sync(counter, bar * NWG);   // h published; lds_h safe to reuse
    }
}

// ---------------------------------------------------------------------------
// Launch
// ---------------------------------------------------------------------------
extern "C" void kernel_launch(void* const* d_in, const int* in_sizes, int n_in,
                              void* d_out, int out_size, void* d_ws, size_t ws_size,
                              hipStream_t stream) {
    const float* src   = (const float*)d_in[0];
    // d_in[1] = tgt (only its length T=100 matters)
    const float* h0    = (const float*)d_in[2];
    const float* c0    = (const float*)d_in[3];
    const float* wih   = (const float*)d_in[4];
    const float* whh   = (const float*)d_in[5];
    const float* bih   = (const float*)d_in[6];
    const float* bhh   = (const float*)d_in[7];
    const float* wpost = (const float*)d_in[8];
    const float* bpost = (const float*)d_in[9];
    float* out = (float*)d_out;

    // workspace layout (~18.1 MB)
    const size_t WSZ = (size_t)Gdim * Hdim * sizeof(__bf16);  // 4,718,592
    const size_t SSZ = (size_t)Bdim * Hdim * sizeof(__bf16);  //   393,216
    char* ws = (char*)d_ws;
    __bf16* wc_b   = (__bf16*)(ws);
    __bf16* wih_b  = (__bf16*)(ws + WSZ);
    __bf16* whh_b  = (__bf16*)(ws + 2 * WSZ);
    __bf16* xbuf   = (__bf16*)(ws + 3 * WSZ);
    __bf16* hbuf   = (__bf16*)(ws + 3 * WSZ + SSZ);
    float*  gates  = (float*) (ws + 3 * WSZ + 2 * SSZ);
    unsigned* cnt  = (unsigned*)(ws + 3 * WSZ + 2 * SSZ + (size_t)Bdim * Gdim * sizeof(float));

    prep_weights<<<2048, 256, 0, stream>>>(wih, whh, wc_b, wih_b, whh_b);
    prep_state<<<768, 256, 0, stream>>>(src, h0, xbuf, hbuf, cnt);
    lstm_decoder<<<NWG, THREADS, SMEM_BYTES, stream>>>(wc_b, wih_b, whh_b, xbuf, hbuf,
                                                       gates, cnt, c0, bih, bhh,
                                                       wpost, bpost, out);
}
